// BaseRNN_61108794687889
// MI455X (gfx1250) — matrix operand order repaired and checked by reading, not compiled, not run on hardware
//
#include <hip/hip_runtime.h>
#include <hip/hip_bf16.h>

// SimpleRNN on MI455X (gfx1250, wave32):
//  Phase 1 (memory-bound): xproj = x @ Wx + b  via v_wmma_f32_16x16x32_f16,
//           f32 accumulate, stored f32 to d_ws ([B*T][64], 64 MB).
//  Phase 2 (latency-bound scan): hT_t = tanh(WhT @ hT_{t-1} + xprojT_t).
//           h kept as the WMMA *B* operand; A=WhT is loop-invariant in VGPRs.
//           A logical-H relabeling sigma(m,v,half)=32*(m>>1)+16*half+8*(m&1)+v
//           makes D-layout -> next-step B-layout a pure same-lane cvt_pk
//           (no LDS, no cross-lane ops in the 4096-step dependence chain).
//           Dense(1)+sigmoid head fused (one shfl_xor(16) reduction per step).

typedef __attribute__((ext_vector_type(16))) _Float16 v16h;
typedef __attribute__((ext_vector_type(8)))  float    v8f;
typedef __attribute__((ext_vector_type(4)))  float    f4;

#define TT 4096
#define BB 64
#define FF 64
#define HH 64

__device__ __forceinline__ float fast_tanh(float x) {
#if __has_builtin(__builtin_amdgcn_tanhf)
  return __builtin_amdgcn_tanhf(x);       // v_tanh_f32 (CDNA5 TRANS op)
#else
  return tanhf(x);
#endif
}

__device__ __forceinline__ float fast_sigmoid(float x) {
#if __has_builtin(__builtin_amdgcn_exp2f) && __has_builtin(__builtin_amdgcn_rcpf)
  float e = __builtin_amdgcn_exp2f(x * -1.442695040888963f);
  return __builtin_amdgcn_rcpf(1.0f + e);
#else
  return 1.0f / (1.0f + __expf(-x));
#endif
}

// ---------------- Phase 1: xproj = x @ Wx + bias ----------------
__global__ void __launch_bounds__(256) rnn_xproj_kernel(
    const float* __restrict__ x, const float* __restrict__ Wx,
    const float* __restrict__ bias, float* __restrict__ xproj)
{
  const int lane = threadIdx.x & 31;
  const int hhalf = lane >> 4;
  const int l16  = lane & 15;
  const int gwave  = blockIdx.x * (blockDim.x >> 5) + (threadIdx.x >> 5);
  const int nwaves = gridDim.x * (blockDim.x >> 5);

  // Wx as B-fragments: lanes<16: V0-3 K=0..7, V4-7 K=16..23; lanes>=16: +8.
  v16h wx[2][4];
  float bb[4];
#pragma unroll
  for (int nt = 0; nt < 4; ++nt) {
    const int N = 16 * nt + l16;
    bb[nt] = bias[N];
#pragma unroll
    for (int c = 0; c < 2; ++c) {
      v16h f;
#pragma unroll
      for (int j = 0; j < 8; ++j) {
        const int K = 32 * c + 16 * (j >> 2) + 8 * hhalf + 2 * (j & 3);
        f[2 * j]     = (_Float16)Wx[(size_t)K * HH + N];
        f[2 * j + 1] = (_Float16)Wx[(size_t)(K + 1) * HH + N];
      }
      wx[c][nt] = f;
    }
  }

  const int ntiles = (BB * TT) / 16;
  for (int tile = gwave; tile < ntiles; tile += nwaves) {
    // A-fragments from x (row = 16*tile + l16). A layout (16-bit, 16x32):
    // lanes<16: V0-3 K=0..7, V4-7 K=16..23; lanes>=16: +8.
    const float* xr = x + (size_t)(16 * tile + l16) * FF;
    v16h a[2];
#pragma unroll
    for (int c = 0; c < 2; ++c) {
      f4 r0 = *(const f4*)(xr + 32 * c + 8 * hhalf);
      f4 r1 = *(const f4*)(xr + 32 * c + 8 * hhalf + 4);
      f4 r2 = *(const f4*)(xr + 32 * c + 16 + 8 * hhalf);
      f4 r3 = *(const f4*)(xr + 32 * c + 16 + 8 * hhalf + 4);
      v16h f;
#pragma unroll
      for (int e = 0; e < 4; ++e) {
        f[e]      = (_Float16)r0[e];
        f[4 + e]  = (_Float16)r1[e];
        f[8 + e]  = (_Float16)r2[e];
        f[12 + e] = (_Float16)r3[e];
      }
      a[c] = f;
    }
#pragma unroll
    for (int nt = 0; nt < 4; ++nt) {
      v8f acc;
#pragma unroll
      for (int v = 0; v < 8; ++v) acc[v] = bb[nt];
      acc = __builtin_amdgcn_wmma_f32_16x16x32_f16(false, a[0], false, wx[0][nt],
                                                   (short)0, acc, false, false);
      acc = __builtin_amdgcn_wmma_f32_16x16x32_f16(false, a[1], false, wx[1][nt],
                                                   (short)0, acc, false, false);
      float* outp = xproj + (size_t)(16 * tile) * HH + 16 * nt + l16;
#pragma unroll
      for (int v = 0; v < 8; ++v)
        outp[(size_t)(v + 8 * hhalf) * HH] = acc[v];
    }
  }
}

// ---------------- Phase 2: sequential scan + fused head ----------------
__global__ void __launch_bounds__(32) rnn_scan_kernel(
    const float* __restrict__ Wh, const float* __restrict__ Wd,
    const float* __restrict__ bdp, const float* __restrict__ xproj,
    float* __restrict__ out)
{
  const int lane  = threadIdx.x & 31;
  const int hhalf = lane >> 4;
  const int l16   = lane & 15;
  const int b     = blockIdx.x * 16 + l16;   // 4 blocks x 16 batch columns
  const float bd  = bdp[0];

  // A = WhT with output-row relabeling sigma; A[M][K] = Wh[K][sigma_out(m,M)]
  v16h A[4][2];
  const int M = l16;
#pragma unroll
  for (int m = 0; m < 4; ++m) {
    const int Hout = 32 * (m >> 1) + 16 * (M >> 3) + 8 * (m & 1) + (M & 7);
#pragma unroll
    for (int c = 0; c < 2; ++c) {
      v16h f;
#pragma unroll
      for (int j = 0; j < 8; ++j) {
        const int K = 32 * c + 16 * hhalf + 2 * j;
        f[2 * j]     = (_Float16)Wh[(size_t)K * HH + Hout];
        f[2 * j + 1] = (_Float16)Wh[(size_t)(K + 1) * HH + Hout];
      }
      A[m][c] = f;
    }
  }

  // Per-lane Dense(1) weights under the same relabeling.
  f4 wd[4][2];
#pragma unroll
  for (int m = 0; m < 4; ++m) {
    const int Hb = 32 * (m >> 1) + 16 * hhalf + 8 * (m & 1);
    wd[m][0] = *(const f4*)(Wd + Hb);
    wd[m][1] = *(const f4*)(Wd + Hb + 4);
  }

  // h0 = 0 -> zero B-fragments
  v16h Bf[2];
#pragma unroll
  for (int c = 0; c < 2; ++c)
#pragma unroll
    for (int e = 0; e < 16; ++e) Bf[c][e] = (_Float16)0.0f;

  const float* xbase = xproj + (size_t)b * TT * HH;
  float* obase = out + (size_t)b * TT;

  for (int t = 0; t < TT; ++t) {
    const float* xr = xbase + (size_t)t * HH;
    if (t + 8 < TT) __builtin_prefetch(xbase + (size_t)(t + 8) * HH, 0, 0);

    // Accumulator init = xprojT fragment (relabeled H).
    v8f acc[4];
#pragma unroll
    for (int m = 0; m < 4; ++m) {
      const int Hb = 32 * (m >> 1) + 16 * hhalf + 8 * (m & 1);
      f4 p0 = *(const f4*)(xr + Hb);
      f4 p1 = *(const f4*)(xr + Hb + 4);
      v8f a;
#pragma unroll
      for (int e = 0; e < 4; ++e) { a[e] = p0[e]; a[4 + e] = p1[e]; }
      acc[m] = a;
    }

    // hT_pre = WhT @ hT + xprojT   (8 WMMAs on the dependence chain)
#pragma unroll
    for (int m = 0; m < 4; ++m) {
      acc[m] = __builtin_amdgcn_wmma_f32_16x16x32_f16(false, A[m][0], false, Bf[0],
                                                      (short)0, acc[m], false, false);
      acc[m] = __builtin_amdgcn_wmma_f32_16x16x32_f16(false, A[m][1], false, Bf[1],
                                                      (short)0, acc[m], false, false);
    }

    // tanh + fused head partial
    float hs[4][8];
    float partial = 0.0f;
#pragma unroll
    for (int m = 0; m < 4; ++m)
#pragma unroll
      for (int v = 0; v < 8; ++v) {
        hs[m][v] = fast_tanh(acc[m][v]);
        partial += hs[m][v] * ((v < 4) ? wd[m][0][v] : wd[m][1][v - 4]);
      }

    // Next-step B fragments: pure same-lane packing thanks to relabeling.
#pragma unroll
    for (int c = 0; c < 2; ++c) {
      v16h f;
#pragma unroll
      for (int j = 0; j < 8; ++j) {
        const int m = 2 * c + (j >> 2);
        const int u = j & 3;
        f[2 * j]     = (_Float16)hs[m][2 * u];
        f[2 * j + 1] = (_Float16)hs[m][2 * u + 1];
      }
      Bf[c] = f;
    }

    // Head: lane L and L^16 hold complementary H halves of column b.
    const float tot = partial + __shfl_xor(partial, 16, 32);
    if (hhalf == 0) obase[t] = fast_sigmoid(tot + bd);
  }
}

extern "C" void kernel_launch(void* const* d_in, const int* in_sizes, int n_in,
                              void* d_out, int out_size, void* d_ws, size_t ws_size,
                              hipStream_t stream) {
  (void)in_sizes; (void)n_in; (void)out_size; (void)ws_size;
  const float* x    = (const float*)d_in[0];
  const float* Wx   = (const float*)d_in[1];
  const float* Wh   = (const float*)d_in[2];
  const float* bias = (const float*)d_in[3];
  const float* Wd   = (const float*)d_in[4];
  const float* bd   = (const float*)d_in[5];
  float* out   = (float*)d_out;
  float* xproj = (float*)d_ws;   // needs B*T*H*4 = 64 MB of workspace

  rnn_xproj_kernel<<<dim3(512), dim3(256), 0, stream>>>(x, Wx, bias, xproj);
  rnn_scan_kernel<<<dim3(4), dim3(32), 0, stream>>>(Wh, Wd, bd, xproj, out);
}
